// GPSLayer_9466107920690
// MI455X (gfx1250) — compile-verified, matrix operand order checked by reading, not run on hardware
//
#include <hip/hip_runtime.h>
#include <hip/hip_bf16.h>
#include <math.h>

// Problem constants (match reference)
#define NN 8192
#define DD 256
#define HH 4
#define DH 64
#define EE 262144

typedef __attribute__((ext_vector_type(16))) __bf16 v16bf;
typedef __attribute__((ext_vector_type(8)))  float  v8f;
typedef __attribute__((ext_vector_type(8)))  unsigned short v8us;

// ---------------------------------------------------------------------------
// helpers
// ---------------------------------------------------------------------------
static __device__ inline unsigned short f2bf(float f) {
  unsigned int u = __float_as_uint(f);
  u += 0x7FFFu + ((u >> 16) & 1u);   // round-to-nearest-even
  return (unsigned short)(u >> 16);
}

static __device__ inline v8f wmma_bf16(v16bf a, v16bf b, v8f c) {
  // D(16x16,f32) = A(16x32,bf16) x B(32x16,bf16) + C
  return __builtin_amdgcn_wmma_f32_16x16x32_bf16(
      /*neg_a=*/false, a, /*neg_b=*/false, b,
      /*c_mod=*/(short)0, c, /*reuse_a=*/false, /*reuse_b=*/false);
}

// Load a 16x32 bf16 fragment (A layout; identical layout serves B when the
// stored matrix is the transpose of B, i.e. W rows = B columns).
// Lane L: row = base_row + (L&15); K chunks [half*8, half*8+8) and
// [16+half*8, ...) where half = L>>4.  16-byte vector loads.
static __device__ inline v16bf load_frag(const unsigned short* base, int stride,
                                         int row, int k0) {
  int lane = threadIdx.x & 31;
  int m = lane & 15, half = lane >> 4;
  const unsigned short* p = base + (size_t)(row + m) * stride + k0 + half * 8;
  union { v8us h[2]; v16bf v; } u;
  u.h[0] = *(const v8us*)(p);
  u.h[1] = *(const v8us*)(p + 16);
  return u.v;
}

// ---------------------------------------------------------------------------
// small utility kernels
// ---------------------------------------------------------------------------
__global__ void zero_u32_kernel(unsigned int* p, size_t n) {
  size_t i = (size_t)blockIdx.x * blockDim.x + threadIdx.x;
  if (i < n) p[i] = 0u;
}

__global__ void f2bf_kernel(const float* __restrict__ in,
                            unsigned short* __restrict__ outp, size_t n) {
  size_t i = (size_t)blockIdx.x * blockDim.x + threadIdx.x;
  if (i < n) outp[i] = f2bf(in[i]);
}

// ---------------------------------------------------------------------------
// sparse adjacency: dedup edges via bitmask (matches .at[].set(1.0) semantics)
// ---------------------------------------------------------------------------
__global__ void edge_mark_kernel(const long long* __restrict__ ei,
                                 unsigned int* __restrict__ mask,
                                 float* __restrict__ deg,
                                 unsigned int* __restrict__ flag) {
  int e = blockIdx.x * blockDim.x + threadIdx.x;
  if (e >= EE) return;
  int s = (int)ei[e];
  int d = (int)ei[EE + e];
  unsigned int bit = 1u << (d & 31);
  unsigned int old = atomicOr(&mask[(size_t)s * (NN / 32) + (d >> 5)], bit);
  unsigned int isNew = ((old & bit) == 0u) ? 1u : 0u;
  flag[e] = isNew;
  if (isNew) atomicAdd(&deg[s], 1.0f);
}

__global__ __launch_bounds__(256) void edge_agg_kernel(
    const long long* __restrict__ ei, const unsigned int* __restrict__ flag,
    const float* __restrict__ x, float* __restrict__ agg) {
  int e = blockIdx.x;                 // one block per edge, 256 dims
  if (!flag[e]) return;
  int s = (int)ei[e];
  int d = (int)ei[EE + e];
  int t = threadIdx.x;
  atomicAdd(&agg[(size_t)s * DD + t], x[(size_t)d * DD + t]);
}

__global__ void finalize_agg_kernel(const float* __restrict__ agg,
                                    const float* __restrict__ deg,
                                    unsigned short* __restrict__ aggb) {
  size_t i = (size_t)blockIdx.x * blockDim.x + threadIdx.x;
  if (i >= (size_t)NN * DD) return;
  int n = (int)(i / DD);
  float inv = 1.0f / fmaxf(deg[n], 1.0f);
  aggb[i] = f2bf(agg[i] * inv);
}

// ---------------------------------------------------------------------------
// WMMA GEMM: out = op(A(M,K) @ W(Nout,K)^T + bias) [+ residual]
// A, W bf16; out f32 and/or bf16.
// One wave -> 16x64 output tile; 2x unrolled K loop with two disjoint
// fragment register sets (no rotation copies). All call sites have an even
// number of 32-wide K tiles (K = 256 or 512), so no remainder handling.
// ---------------------------------------------------------------------------
__global__ __launch_bounds__(128) void gemm_bf16_kernel(
    const unsigned short* __restrict__ A, const unsigned short* __restrict__ W,
    const float* __restrict__ bias, const float* __restrict__ resid,
    float* __restrict__ outf, unsigned short* __restrict__ outb,
    int M, int Nout, int K, int doGelu) {
  int wave = threadIdx.x >> 5;
  int tm = blockIdx.x;
  int tc0 = (blockIdx.y * 4 + wave) * 64;       // first output column of wave
  if (tc0 >= Nout) return;
  int rowA = tm * 16;
  v8f z8 = {0.f, 0.f, 0.f, 0.f, 0.f, 0.f, 0.f, 0.f};
  v8f acc0 = z8, acc1 = z8, acc2 = z8, acc3 = z8;
  int kTiles = K >> 5;                          // even (>= 8) at all call sites

  // set 0: fragments for kt = 0
  v16bf a0  = load_frag(A, K, rowA, 0);
  v16bf b00 = load_frag(W, K, tc0 + 0,  0);
  v16bf b01 = load_frag(W, K, tc0 + 16, 0);
  v16bf b02 = load_frag(W, K, tc0 + 32, 0);
  v16bf b03 = load_frag(W, K, tc0 + 48, 0);

  int kt = 0;
  for (; kt + 2 < kTiles; kt += 2) {
    int k1 = (kt + 1) * 32;
    v16bf a1  = load_frag(A, K, rowA, k1);      // set 1 loads (unconditional)
    v16bf b10 = load_frag(W, K, tc0 + 0,  k1);
    v16bf b11 = load_frag(W, K, tc0 + 16, k1);
    v16bf b12 = load_frag(W, K, tc0 + 32, k1);
    v16bf b13 = load_frag(W, K, tc0 + 48, k1);
    __builtin_prefetch(A + (size_t)rowA * K + (size_t)(kt + 2) * 32, 0, 1);
    __builtin_prefetch(W + (size_t)tc0 * K + (size_t)(kt + 2) * 32, 0, 1);

    acc0 = wmma_bf16(a0, b00, acc0);            // consume set 0
    acc1 = wmma_bf16(a0, b01, acc1);
    acc2 = wmma_bf16(a0, b02, acc2);
    acc3 = wmma_bf16(a0, b03, acc3);

    int k2 = (kt + 2) * 32;
    a0  = load_frag(A, K, rowA, k2);            // refill set 0 (unconditional)
    b00 = load_frag(W, K, tc0 + 0,  k2);
    b01 = load_frag(W, K, tc0 + 16, k2);
    b02 = load_frag(W, K, tc0 + 32, k2);
    b03 = load_frag(W, K, tc0 + 48, k2);

    acc0 = wmma_bf16(a1, b10, acc0);            // consume set 1
    acc1 = wmma_bf16(a1, b11, acc1);
    acc2 = wmma_bf16(a1, b12, acc2);
    acc3 = wmma_bf16(a1, b13, acc3);
  }
  {                                             // final pair, no lookahead
    int k1 = (kt + 1) * 32;
    v16bf a1  = load_frag(A, K, rowA, k1);
    v16bf b10 = load_frag(W, K, tc0 + 0,  k1);
    v16bf b11 = load_frag(W, K, tc0 + 16, k1);
    v16bf b12 = load_frag(W, K, tc0 + 32, k1);
    v16bf b13 = load_frag(W, K, tc0 + 48, k1);
    acc0 = wmma_bf16(a0, b00, acc0);
    acc1 = wmma_bf16(a0, b01, acc1);
    acc2 = wmma_bf16(a0, b02, acc2);
    acc3 = wmma_bf16(a0, b03, acc3);
    acc0 = wmma_bf16(a1, b10, acc0);
    acc1 = wmma_bf16(a1, b11, acc1);
    acc2 = wmma_bf16(a1, b12, acc2);
    acc3 = wmma_bf16(a1, b13, acc3);
  }

  int lane = threadIdx.x & 31;
  int nn = lane & 15, half = lane >> 4;
  v8f accs[4] = {acc0, acc1, acc2, acc3};
#pragma unroll
  for (int c = 0; c < 4; ++c) {
    int col = tc0 + c * 16 + nn;
    float bv = bias ? bias[col] : 0.0f;
#pragma unroll
    for (int r = 0; r < 8; ++r) {
      int row = rowA + half * 8 + r;
      float v = accs[c][r] + bv;
      if (doGelu) v = 0.5f * v * (1.0f + erff(v * 0.70710678118654752f));
      if (resid) v += resid[(size_t)row * Nout + col];
      size_t o = (size_t)row * Nout + col;
      if (outf) outf[o] = v;
      if (outb) outb[o] = f2bf(v);
    }
  }
}

// ---------------------------------------------------------------------------
// split qkv (f32, N x 3D) into per-head bf16 Q (pre-scaled), K, and V^T
// ---------------------------------------------------------------------------
__global__ void qkv_split_kernel(const float* __restrict__ qkv,
                                 unsigned short* __restrict__ qb,
                                 unsigned short* __restrict__ kb,
                                 unsigned short* __restrict__ vtb) {
  size_t i = (size_t)blockIdx.x * blockDim.x + threadIdx.x;
  if (i >= (size_t)NN * DD) return;
  int n = (int)(i / DD);
  int d = (int)(i % DD);
  int h = d >> 6, dd = d & 63;
  const float* row = qkv + (size_t)n * (3 * DD);
  qb[((size_t)h * NN + n) * DH + dd] = f2bf(row[d] * 0.125f);   // 1/sqrt(64)
  kb[((size_t)h * NN + n) * DH + dd] = f2bf(row[DD + d]);
  vtb[((size_t)h * DH + dd) * NN + n] = f2bf(row[2 * DD + d]);  // transposed
}

// ---------------------------------------------------------------------------
// Flash attention: one wave per (16-query tile, head). Streams 32-key tiles:
// 4 WMMAs for S = QK^T, online softmax via LDS, 4 WMMAs for O += P V.
// One step of the online-softmax loop; consumes the current K fragments and
// unconditionally loads the next tile's K fragments into the other register
// set (K buffer is padded by 32 rows so the final lookahead stays in-bounds).
// ---------------------------------------------------------------------------
static __device__ __forceinline__ void flash_step(
    int kb0, v16bf q0, v16bf q1,
    v16bf k00, v16bf k01, v16bf k10, v16bf k11,
    v16bf& n00, v16bf& n01, v16bf& n10, v16bf& n11,
    const unsigned short* __restrict__ Kh, const unsigned short* __restrict__ Vt,
    v8f& acc0, v8f& acc1, v8f& acc2, v8f& acc3,
    float& mrow, float& lrow,
    float (*sS)[32], unsigned short (*sP)[32], float* sF) {
  const v8f zero8 = {0.f, 0.f, 0.f, 0.f, 0.f, 0.f, 0.f, 0.f};
  int lane = threadIdx.x & 31;
  int nn = lane & 15, half = lane >> 4;

  // S(16x32) = Q(16x64) @ K^T(64x32)
  v8f s0 = zero8, s1 = zero8;
  s0 = wmma_bf16(q0, k00, s0);
  s0 = wmma_bf16(q1, k01, s0);
  s1 = wmma_bf16(q0, k10, s1);
  s1 = wmma_bf16(q1, k11, s1);

  // V fragments for this tile; in flight during the softmax section
  v16bf v0 = load_frag(Vt, NN, 0,  kb0);
  v16bf v1 = load_frag(Vt, NN, 16, kb0);
  v16bf v2 = load_frag(Vt, NN, 32, kb0);
  v16bf v3 = load_frag(Vt, NN, 48, kb0);

  // next tile's K fragments (unconditional; padded buffer)
  int kb1 = kb0 + 32;
  n00 = load_frag(Kh, DH, kb1, 0);
  n01 = load_frag(Kh, DH, kb1, 32);
  n10 = load_frag(Kh, DH, kb1 + 16, 0);
  n11 = load_frag(Kh, DH, kb1 + 16, 32);

  // C layout -> LDS row-major score tile
#pragma unroll
  for (int r = 0; r < 8; ++r) {
    sS[half * 8 + r][nn] = s0[r];
    sS[half * 8 + r][16 + nn] = s1[r];
  }
  __syncthreads();

  // online softmax: lane = query row
  if (lane < 16) {
    float rmax = mrow;
#pragma unroll 4
    for (int j = 0; j < 32; ++j) rmax = fmaxf(rmax, sS[lane][j]);
    float f = __expf(mrow - rmax);
    float sum = 0.0f;
#pragma unroll 4
    for (int j = 0; j < 32; ++j) {
      float pv = __expf(sS[lane][j] - rmax);
      sP[lane][j] = f2bf(pv);
      sum += pv;
    }
    lrow = lrow * f + sum;
    mrow = rmax;
    sF[lane] = f;
  }
  __syncthreads();

  // rescale running accumulator by per-row factor
#pragma unroll
  for (int r = 0; r < 8; ++r) {
    float f = sF[half * 8 + r];
    acc0[r] *= f; acc1[r] *= f; acc2[r] *= f; acc3[r] *= f;
  }

  // O(16x64) += P(16x32) @ V(32x64); V^T rows give contiguous B columns
  v16bf pf = load_frag(&sP[0][0], 32, 0, 0);
  acc0 = wmma_bf16(pf, v0, acc0);
  acc1 = wmma_bf16(pf, v1, acc1);
  acc2 = wmma_bf16(pf, v2, acc2);
  acc3 = wmma_bf16(pf, v3, acc3);
  __syncthreads();
}

__global__ __launch_bounds__(32) void flash_attn_kernel(
    const unsigned short* __restrict__ qb, const unsigned short* __restrict__ kb,
    const unsigned short* __restrict__ vtb, unsigned short* __restrict__ attnb) {
  __shared__ __align__(16) float sS[16][32];
  __shared__ __align__(16) unsigned short sP[16][32];
  __shared__ float sF[16];
  __shared__ float sR[16];

  int t = blockIdx.x, h = blockIdx.y;
  int lane = threadIdx.x & 31;
  int nn = lane & 15, half = lane >> 4;
  const unsigned short* Qh = qb + (size_t)h * NN * DH;
  const unsigned short* Kh = kb + (size_t)h * NN * DH;
  const unsigned short* Vt = vtb + (size_t)h * DH * NN;

  v16bf q0 = load_frag(Qh, DH, t * 16, 0);
  v16bf q1 = load_frag(Qh, DH, t * 16, 32);

  v8f zero8 = {0.f, 0.f, 0.f, 0.f, 0.f, 0.f, 0.f, 0.f};
  v8f acc0 = zero8, acc1 = zero8, acc2 = zero8, acc3 = zero8;
  float mrow = -3.0e38f, lrow = 0.0f;   // live in lanes 0..15 (row = lane)

  // set A: K fragments for the first key tile
  v16bf ka0 = load_frag(Kh, DH, 0, 0);
  v16bf ka1 = load_frag(Kh, DH, 0, 32);
  v16bf ka2 = load_frag(Kh, DH, 16, 0);
  v16bf ka3 = load_frag(Kh, DH, 16, 32);
  v16bf kb0f = ka0, kb1f = ka1, kb2f = ka2, kb3f = ka3;  // set B

  // 2x unrolled: phase A consumes set A / fills set B, phase B vice versa.
  for (int kt = 0; kt < NN / 32; kt += 2) {
    flash_step(kt * 32, q0, q1, ka0, ka1, ka2, ka3,
               kb0f, kb1f, kb2f, kb3f, Kh, Vt,
               acc0, acc1, acc2, acc3, mrow, lrow, sS, sP, sF);
    flash_step((kt + 1) * 32, q0, q1, kb0f, kb1f, kb2f, kb3f,
               ka0, ka1, ka2, ka3, Kh, Vt,
               acc0, acc1, acc2, acc3, mrow, lrow, sS, sP, sF);
  }

  if (lane < 16) sR[lane] = 1.0f / lrow;
  __syncthreads();
#pragma unroll
  for (int r = 0; r < 8; ++r) {
    int row = t * 16 + half * 8 + r;
    float inv = sR[half * 8 + r];
    size_t base = (size_t)row * DD + (size_t)h * DH + nn;
    attnb[base + 0]  = f2bf(acc0[r] * inv);
    attnb[base + 16] = f2bf(acc1[r] * inv);
    attnb[base + 32] = f2bf(acc2[r] * inv);
    attnb[base + 48] = f2bf(acc3[r] * inv);
  }
}

// ---------------------------------------------------------------------------
// host orchestration
// ---------------------------------------------------------------------------
extern "C" void kernel_launch(void* const* d_in, const int* in_sizes, int n_in,
                              void* d_out, int out_size, void* d_ws, size_t ws_size,
                              hipStream_t stream) {
  const float* x        = (const float*)d_in[0];
  const long long* ei   = (const long long*)d_in[1];   // int64 edge_index (2,E)
  const float* Wl       = (const float*)d_in[2];
  const float* bl       = (const float*)d_in[3];
  const float* Win      = (const float*)d_in[4];
  const float* bin      = (const float*)d_in[5];
  const float* Wo       = (const float*)d_in[6];
  const float* bo       = (const float*)d_in[7];
  const float* W1       = (const float*)d_in[8];
  const float* b1       = (const float*)d_in[9];
  const float* W2       = (const float*)d_in[10];
  const float* b2       = (const float*)d_in[11];
  float* out = (float*)d_out;

  // workspace carve-out (~99 MB total)
  char* p = (char*)d_ws;
  auto take = [&](size_t bytes) -> char* {
    char* r = p;
    p += (bytes + 255) & ~(size_t)255;
    return r;
  };
  unsigned int*   mask  = (unsigned int*)  take((size_t)NN * (NN / 32) * 4);
  float*          deg   = (float*)         take((size_t)NN * 4);
  float*          agg   = (float*)         take((size_t)NN * DD * 4);
  unsigned int*   flag  = (unsigned int*)  take((size_t)EE * 4);
  unsigned short* aggb  = (unsigned short*)take((size_t)NN * DD * 2);
  float*          x1    = (float*)         take((size_t)NN * DD * 4);
  unsigned short* xb    = (unsigned short*)take((size_t)NN * DD * 2);
  float*          qkvf  = (float*)         take((size_t)NN * 3 * DD * 4);
  unsigned short* qbuf  = (unsigned short*)take((size_t)NN * DD * 2);
  // kbuf padded by 32 rows (dh=64) for the flash kernel's lookahead load
  unsigned short* kbuf  = (unsigned short*)take((size_t)NN * DD * 2 + 32 * DH * 2);
  unsigned short* vtbuf = (unsigned short*)take((size_t)NN * DD * 2);
  unsigned short* attnb = (unsigned short*)take((size_t)NN * DD * 2);
  float*          x2    = (float*)         take((size_t)NN * DD * 4);
  unsigned short* x2b   = (unsigned short*)take((size_t)NN * DD * 2);
  unsigned short* hb    = (unsigned short*)take((size_t)NN * 2 * DD * 2);
  unsigned short* wlb   = (unsigned short*)take((size_t)DD * DD * 2);
  unsigned short* winb  = (unsigned short*)take((size_t)3 * DD * DD * 2);
  unsigned short* wob   = (unsigned short*)take((size_t)DD * DD * 2);
  unsigned short* w1b   = (unsigned short*)take((size_t)2 * DD * DD * 2);
  unsigned short* w2b   = (unsigned short*)take((size_t)DD * 2 * DD * 2);

  // 1) clear adjacency bitmask / degree / aggregation accumulators
  size_t maskW = (size_t)NN * (NN / 32);
  zero_u32_kernel<<<(unsigned)((maskW + 255) / 256), 256, 0, stream>>>(mask, maskW);
  zero_u32_kernel<<<(NN + 255) / 256, 256, 0, stream>>>((unsigned int*)deg, NN);
  zero_u32_kernel<<<(unsigned)(((size_t)NN * DD + 255) / 256), 256, 0, stream>>>(
      (unsigned int*)agg, (size_t)NN * DD);

  // 2) weight conversions to bf16
  f2bf_kernel<<<(DD * DD + 255) / 256, 256, 0, stream>>>(Wl, wlb, (size_t)DD * DD);
  f2bf_kernel<<<(3 * DD * DD + 255) / 256, 256, 0, stream>>>(Win, winb, (size_t)3 * DD * DD);
  f2bf_kernel<<<(DD * DD + 255) / 256, 256, 0, stream>>>(Wo, wob, (size_t)DD * DD);
  f2bf_kernel<<<(2 * DD * DD + 255) / 256, 256, 0, stream>>>(W1, w1b, (size_t)2 * DD * DD);
  f2bf_kernel<<<(DD * 2 * DD + 255) / 256, 256, 0, stream>>>(W2, w2b, (size_t)DD * 2 * DD);

  // 3) sparse local aggregation (deduped, row-normalized)
  edge_mark_kernel<<<(EE + 255) / 256, 256, 0, stream>>>(ei, mask, deg, flag);
  edge_agg_kernel<<<EE, 256, 0, stream>>>(ei, flag, x, agg);
  finalize_agg_kernel<<<(unsigned)(((size_t)NN * DD + 255) / 256), 256, 0, stream>>>(
      agg, deg, aggb);

  // 4) x1 = x + aggn @ W_local^T + b_local   (also emits bf16 copy xb)
  gemm_bf16_kernel<<<dim3(NN / 16, 1), 128, 0, stream>>>(
      aggb, wlb, bl, x, x1, xb, NN, DD, DD, 0);

  // 5) qkv = x1 @ W_in^T + b_in
  gemm_bf16_kernel<<<dim3(NN / 16, 3), 128, 0, stream>>>(
      xb, winb, bin, nullptr, qkvf, nullptr, NN, 3 * DD, DD, 0);

  // 6) per-head bf16 Q (pre-scaled), K, V^T
  qkv_split_kernel<<<(unsigned)(((size_t)NN * DD + 255) / 256), 256, 0, stream>>>(
      qkvf, qbuf, kbuf, vtbuf);

  // 7) flash attention -> attnb (bf16, N x D)
  flash_attn_kernel<<<dim3(NN / 16, HH), 32, 0, stream>>>(qbuf, kbuf, vtbuf, attnb);

  // 8) x2 = x1 + attn @ W_out^T + b_out
  gemm_bf16_kernel<<<dim3(NN / 16, 1), 128, 0, stream>>>(
      attnb, wob, bo, x1, x2, x2b, NN, DD, DD, 0);

  // 9) h = gelu(x2 @ W1^T + b1)   (bf16 output only)
  gemm_bf16_kernel<<<dim3(NN / 16, 2), 128, 0, stream>>>(
      x2b, w1b, b1, nullptr, nullptr, hb, NN, 2 * DD, DD, 1);

  // 10) out = x2 + h @ W2^T + b2
  gemm_bf16_kernel<<<dim3(NN / 16, 1), 128, 0, stream>>>(
      hb, w2b, b2, x2, out, nullptr, NN, DD, 2 * DD, 0);
}
